// PreciseADR_RGCN_64957085384738
// MI455X (gfx1250) — matrix operand order repaired and checked by reading, not compiled, not run on hardware
//
#include <hip/hip_runtime.h>
#include <hip/hip_bf16.h>
#include <math.h>

// ---------------------------------------------------------------------------
// PreciseADR hetero-GraphSAGE for MI455X (gfx1250, wave32, WMMA).
//  - All dense GEMMs via v_wmma_f32_16x16x32_f16 (f16 in, f32 accumulate).
//  - Edge aggregation (the bandwidth roofline: ~6.5 GB/layer) gathers f16
//    activations and accumulates with global_atomic_add_f32.
// ---------------------------------------------------------------------------

typedef __attribute__((ext_vector_type(16))) _Float16 v16h;
typedef __attribute__((ext_vector_type(8)))  _Float16 v8h;
typedef __attribute__((ext_vector_type(4)))  _Float16 v4h;
typedef __attribute__((ext_vector_type(8)))  float    v8f;
typedef __attribute__((ext_vector_type(4)))  float    v4f;

#define NPAT 200000
#define NDRG 5000
#define NADR 3000
#define IN_D 256
#define H_D  128
#define OUT_D 64

// ---------------- elementwise helpers ----------------
__global__ void cvt_f32_f16(const float* __restrict__ s, _Float16* __restrict__ d, int n) {
  int i = blockIdx.x * blockDim.x + threadIdx.x;
  if (i < n) d[i] = (_Float16)s[i];
}

__global__ void finalize_mean_kernel(const float* __restrict__ agg, const float* __restrict__ cnt,
                                     _Float16* __restrict__ o, int n) {
  int i = blockIdx.x * blockDim.x + threadIdx.x;
  if (i < n) {
    float c = cnt[i >> 7];             // H_D == 128 features per node
    c = c < 1.f ? 1.f : c;
    o[i] = (_Float16)(agg[i] / c);
  }
}

// One wave per edge: lane covers 4 of 128 features (8B f16 gather -> 4 f32 atomics)
__global__ void scatter_accum_kernel(const _Float16* __restrict__ xsrc,
                                     const int* __restrict__ src, const int* __restrict__ dst,
                                     float* __restrict__ agg, float* __restrict__ cnt, int nE) {
  int lane = threadIdx.x & 31;
  int wid  = (blockIdx.x * blockDim.x + threadIdx.x) >> 5;
  int nw   = (gridDim.x * blockDim.x) >> 5;
  for (int e = wid; e < nE; e += nw) {
    int s = src[e], d = dst[e];
    v4h x = *(const v4h*)(xsrc + (size_t)s * H_D + lane * 4);
    float* ap = agg + (size_t)d * H_D + lane * 4;
    atomicAdd(ap + 0, (float)x[0]);
    atomicAdd(ap + 1, (float)x[1]);
    atomicAdd(ap + 2, (float)x[2]);
    atomicAdd(ap + 3, (float)x[3]);
    if (lane == 0) atomicAdd(cnt + d, 1.0f);
  }
}

// ---------------- WMMA GEMM: C[M,N] = A[M,K] * W[N,K]^T (+bias/accum) ----------------
// Wave owns a 16-row slab and all NT 16-col tiles (A fragment reused NT times).
// Fragment layouts per CDNA5 ISA 7.12.2:
//   A lane l: row m=l%16; element e holds k = 16*(e>=8) + 8*(l>=16) + (e%8)
//             -> two contiguous 8-half (16B) chunks per lane.
//   B lane l: col n=l%16;   element e holds k = 16*(e>=8) + 8*(l>=16) + (e%8)
//             -> two contiguous 8-half (16B) chunks of row n of W[N,K].
//   C lane l: col n=l%16, vgpr r: row m = r + 8*(l>=16).
// CMODE: 1=bias init, 2=load C, 3=load C + bias.  EPI: 0=f32 store, 1=tanh->f16.
template<int NT, int KSTEPS, int CMODE, int EPI, bool A32>
__global__ __launch_bounds__(256)
void gemm_wmma_kernel(const void* __restrict__ Av, const _Float16* __restrict__ W,
                      const float* __restrict__ bias,
                      float* __restrict__ C32, _Float16* __restrict__ C16, int M) {
  constexpr int K = KSTEPS * 32;
  constexpr int N = NT * 16;
  const int lane = threadIdx.x & 31;
  const int wave = threadIdx.x >> 5;
  const int row0 = (blockIdx.x * 8 + wave) * 16;
  if (row0 >= M) return;                     // wave-uniform exit
  const int hi = lane >> 4;
  const int ln = lane & 15;
  int mA = row0 + ln; if (mA >= M) mA = M - 1;   // clamp OOB rows (stores predicated)

  v8f acc[NT];
#pragma unroll
  for (int t = 0; t < NT; ++t) {
    float bv = 0.f;
    if constexpr (CMODE == 1 || CMODE == 3) bv = bias[t * 16 + ln];
#pragma unroll
    for (int r = 0; r < 8; ++r) {
      float cv = 0.f;
      if constexpr (CMODE >= 2) {
        int m = row0 + r + 8 * hi; if (m >= M) m = M - 1;
        cv = C32[(size_t)m * N + t * 16 + ln];
      }
      acc[t][r] = cv + bv;
    }
  }

#pragma unroll
  for (int ks = 0; ks < KSTEPS; ++ks) {
    const int k0 = ks * 32;
    v16h a;
    if constexpr (A32) {                     // f32 source, convert on the fly
      const float* A = (const float*)Av;
      const v4f* p0 = (const v4f*)(A + (size_t)mA * K + k0 + 8 * hi);
      const v4f* p1 = (const v4f*)(A + (size_t)mA * K + k0 + 16 + 8 * hi);
      v4f f0 = p0[0], f1 = p0[1], f2 = p1[0], f3 = p1[1];
#pragma unroll
      for (int e = 0; e < 4; ++e) {
        a[e]      = (_Float16)f0[e];
        a[4 + e]  = (_Float16)f1[e];
        a[8 + e]  = (_Float16)f2[e];
        a[12 + e] = (_Float16)f3[e];
      }
    } else {
      const _Float16* A = (const _Float16*)Av;
      const _Float16* p = A + (size_t)mA * K + k0 + 8 * hi;
      union { v16h v; v8h h[2]; } u;
      u.h[0] = *(const v8h*)p;
      u.h[1] = *(const v8h*)(p + 16);
      a = u.v;
    }
#pragma unroll
    for (int t = 0; t < NT; ++t) {
      const _Float16* pb = W + (size_t)(t * 16 + ln) * K + k0 + 8 * hi;
      union { v16h v; v8h h[2]; } ub;
      ub.h[0] = *(const v8h*)pb;
      ub.h[1] = *(const v8h*)(pb + 16);
      const v16h b = ub.v;
      acc[t] = __builtin_amdgcn_wmma_f32_16x16x32_f16(false, a, false, b,
                                                      (short)0, acc[t], false, false);
    }
  }

#pragma unroll
  for (int t = 0; t < NT; ++t) {
#pragma unroll
    for (int r = 0; r < 8; ++r) {
      int m = row0 + r + 8 * hi;
      if (m < M) {
        size_t idx = (size_t)m * N + t * 16 + ln;
        if constexpr (EPI == 0)      C32[idx] = acc[t][r];
        else                         C16[idx] = (_Float16)tanhf(acc[t][r]);
      }
    }
  }
}

// ---------------------------------------------------------------------------
extern "C" void kernel_launch(void* const* d_in, const int* in_sizes, int n_in,
                              void* d_out, int out_size, void* d_ws, size_t ws_size,
                              hipStream_t stream) {
  (void)out_size;
  // --- inputs (setup_inputs dict order) ---
  const float* x_in[3] = { (const float*)d_in[0], (const float*)d_in[1], (const float*)d_in[2] };
  const int* e_src[4] = { (const int*)d_in[3], (const int*)d_in[5], (const int*)d_in[7], (const int*)d_in[9]  };
  const int* e_dst[4] = { (const int*)d_in[4], (const int*)d_in[6], (const int*)d_in[8], (const int*)d_in[10] };
  const int  e_cnt[4] = { in_sizes[3], in_sizes[5], in_sizes[7], in_sizes[9] };
  const int  e_sT[4]  = { 0, 1, 0, 2 };  // pd, dp, pa, ap : src node type
  const int  e_dT[4]  = { 1, 0, 2, 0 };  //                  dst node type
  const int  NN[3]    = { NPAT, NDRG, NADR };

  // --- params: either 28 separate tensors after the 11 graph inputs, or one flat blob ---
  const float *in_W, *in_b, *ro_W, *ro_b, *Wl[8], *bl[8], *Wr[8];
  if (n_in >= 39) {
    int i = 11;
    in_W = (const float*)d_in[i++]; in_b = (const float*)d_in[i++];
    ro_W = (const float*)d_in[i++]; ro_b = (const float*)d_in[i++];
    for (int j = 0; j < 8; ++j) {
      Wl[j] = (const float*)d_in[i++]; bl[j] = (const float*)d_in[i++]; Wr[j] = (const float*)d_in[i++];
    }
  } else {
    const float* base = (const float*)d_in[11];
    size_t o = 0;
    in_W = base + o; o += (size_t)H_D * IN_D;
    in_b = base + o; o += H_D;
    ro_W = base + o; o += (size_t)OUT_D * H_D;
    ro_b = base + o; o += OUT_D;
    for (int j = 0; j < 8; ++j) {
      Wl[j] = base + o; o += (size_t)H_D * H_D;
      bl[j] = base + o; o += H_D;
      Wr[j] = base + o; o += (size_t)H_D * H_D;
    }
  }

  // --- workspace carve (f16 activations, f32 accumulators, f16 weights) ---
  auto aup = [](size_t x) { return (x + 255) & ~(size_t)255; };
  char* p = (char*)d_ws;
  _Float16* x16[3]; float* out32[3];
  x16[0]   = (_Float16*)p; p += aup((size_t)NPAT * H_D * 2);
  x16[1]   = (_Float16*)p; p += aup((size_t)NDRG * H_D * 2);
  x16[2]   = (_Float16*)p; p += aup((size_t)NADR * H_D * 2);
  out32[0] = (float*)p;    p += aup((size_t)NPAT * H_D * 4);
  out32[1] = (float*)p;    p += aup((size_t)NDRG * H_D * 4);
  out32[2] = (float*)p;    p += aup((size_t)NADR * H_D * 4);
  float*    agg32 = (float*)p;    p += aup((size_t)NPAT * H_D * 4);
  _Float16* agg16 = (_Float16*)p; p += aup((size_t)NPAT * H_D * 2);
  float*    cnt   = (float*)p;    p += aup((size_t)NPAT * 4);
  _Float16* wf    = (_Float16*)p; p += aup((size_t)(H_D * IN_D + OUT_D * H_D + 16 * H_D * H_D) * 2);
  if ((size_t)(p - (char*)d_ws) > ws_size) return;   // insufficient scratch

  _Float16* in_W16 = wf;
  _Float16* ro_W16 = wf + (size_t)H_D * IN_D;
  _Float16 *Wl16[8], *Wr16[8];
  {
    size_t o = (size_t)H_D * IN_D + (size_t)OUT_D * H_D;
    for (int j = 0; j < 8; ++j) { Wl16[j] = wf + o; o += (size_t)H_D * H_D;
                                  Wr16[j] = wf + o; o += (size_t)H_D * H_D; }
  }

  auto cvt = [&](const float* s, _Float16* d, int n) {
    cvt_f32_f16<<<(n + 255) / 256, 256, 0, stream>>>(s, d, n);
  };
  auto ggrid = [](int M) { return dim3((unsigned)(((M + 15) / 16 + 7) / 8)); };

  // --- weights -> f16 (row-major [N,K]: B-fragment loads are contiguous) ---
  cvt(in_W, in_W16, H_D * IN_D);
  cvt(ro_W, ro_W16, OUT_D * H_D);
  for (int j = 0; j < 8; ++j) { cvt(Wl[j], Wl16[j], H_D * H_D); cvt(Wr[j], Wr16[j], H_D * H_D); }

  // --- input linear + tanh: x16[t] = tanh(x_in @ in_W.T + in_b) ---
  for (int t = 0; t < 3; ++t)
    gemm_wmma_kernel<8, 8, 1, 1, true><<<ggrid(NN[t]), 256, 0, stream>>>(
        x_in[t], in_W16, in_b, nullptr, x16[t], NN[t]);

  // --- 2 hetero-SAGE layers ---
  for (int l = 0; l < 2; ++l) {
    bool first[3] = { true, true, true };
    for (int ei = 0; ei < 4; ++ei) {
      const int dT = e_dT[ei], sT = e_sT[ei], dN = NN[dT];
      const int j  = l * 4 + ei;
      const int nE = e_cnt[ei];
      hipMemsetAsync(agg32, 0, (size_t)dN * H_D * 4, stream);
      hipMemsetAsync(cnt,   0, (size_t)dN * 4, stream);
      int blocks = (nE / 8 < 8192) ? ((nE + 7) / 8) : 8192;
      scatter_accum_kernel<<<blocks, 256, 0, stream>>>(x16[sT], e_src[ei], e_dst[ei], agg32, cnt, nE);
      int nEl = dN * H_D;
      finalize_mean_kernel<<<(nEl + 255) / 256, 256, 0, stream>>>(agg32, cnt, agg16, nEl);
      // out_dst (+)= mean_agg @ Wl.T + bl   (HeteroConv 'sum' merge via C-init mode)
      if (first[dT])
        gemm_wmma_kernel<8, 4, 1, 0, false><<<ggrid(dN), 256, 0, stream>>>(
            agg16, Wl16[j], bl[j], out32[dT], nullptr, dN);
      else
        gemm_wmma_kernel<8, 4, 3, 0, false><<<ggrid(dN), 256, 0, stream>>>(
            agg16, Wl16[j], bl[j], out32[dT], nullptr, dN);
      // out_dst += x_dst @ Wr.T   (no bias)
      gemm_wmma_kernel<8, 4, 2, 0, false><<<ggrid(dN), 256, 0, stream>>>(
          x16[dT], Wr16[j], nullptr, out32[dT], nullptr, dN);
      first[dT] = false;
    }
    for (int t = 0; t < 3; ++t)       // next-layer activations (no nonlinearity between layers)
      cvt(out32[t], x16[t], NN[t] * H_D);
  }

  // --- readout: d_out = x_patient @ ro_W.T + ro_b  (f32 [200000,64]) ---
  gemm_wmma_kernel<4, 4, 1, 0, false><<<ggrid(NPAT), 256, 0, stream>>>(
      x16[0], ro_W16, ro_b, (float*)d_out, nullptr, NPAT);
}
